// FastSpeech2ConformerEncoderLayer_75995151335635
// MI455X (gfx1250) — compile-verified, matrix operand order checked by reading, not run on hardware
//
#include <hip/hip_runtime.h>
#include <hip/hip_bf16.h>
#include <stdint.h>

#define DEV static __device__ __forceinline__

typedef __attribute__((ext_vector_type(16))) __bf16 v16bf;
typedef __attribute__((ext_vector_type(8)))  float  v8f;

union Frag { v16bf v; uint4 q[2]; };

static constexpr int Bc  = 4;
static constexpr int Tc  = 2048;
static constexpr int Dc  = 384;
static constexpr int Hn  = 4;
static constexpr int DHc = 96;
static constexpr int Fc  = 1536;
static constexpr int Mc  = Bc * Tc;        // 8192
static constexpr int Lc  = 2 * Tc - 1;     // 4095

DEV unsigned short f2bf(float f) {
  unsigned int u = __float_as_uint(f);
  u += 0x7FFFu + ((u >> 16) & 1u);   // round-to-nearest-even
  return (unsigned short)(u >> 16);
}

DEV uint4 ldg16(const unsigned short* p) { return *(const uint4*)p; }

DEV v8f wmma_bf16(const Frag& a, const Frag& b, v8f c) {
  return __builtin_amdgcn_wmma_f32_16x16x32_bf16(false, a.v, false, b.v,
                                                 (short)0, c, false, false);
}

// CDNA5 async global->LDS copy (tracked by ASYNCcnt). LDS dest address is the
// low 32 bits of the flat pointer (ISA 10.2).
DEV void async_ld_b128(unsigned lds_off, unsigned long long gaddr) {
  asm volatile("global_load_async_to_lds_b128 %0, %1, off"
               :: "v"(lds_off), "v"(gaddr) : "memory");
}
DEV unsigned lds_addr32(const void* p) {
  return (unsigned)(uintptr_t)p;
}

// ---------------------------------------------------------------------------
// Generic bf16 WMMA GEMM:  C(M,N) = A(M,K) * W(N,K)^T [+ bias] [+= if accum]
// Optional row-shift within batches of T rows (for k=3 temporal convs):
// source row = m + shift, zeroed when (m%T)+shift is out of [0,T).
// Block tile 64(M) x 128(N), 8 waves, each wave a 32x32 tile, K step 32.
// Double-buffered LDS filled by GLOBAL_LOAD_ASYNC_TO_LDS_B128: each wave
// issues exactly 3 async ops per k-tile, so s_wait_asynccnt 3 releases the
// current tile while the next tile's loads remain in flight.
// ---------------------------------------------------------------------------
__global__ __launch_bounds__(256) void gemm_bf16_kernel(
    const unsigned short* __restrict__ A, const unsigned short* __restrict__ W,
    const float* __restrict__ bias, float* __restrict__ C,
    int M, int N, int K, int T, int shift, int accum)
{
  __shared__ __align__(16) unsigned short sA[2][64 * 32];
  __shared__ __align__(16) unsigned short sW[2][128 * 32];

  int m0 = blockIdx.x * 64;
  int n0 = blockIdx.y * 128;
  int tid  = threadIdx.x;
  int lane = tid & 31;
  int wave = tid >> 5;
  int wm = wave >> 2, wn = wave & 3;
  int ln16 = lane & 15;
  int hf   = (lane >> 4) & 1;
  int klo  = hf * 8;

  int arow   = tid >> 2;          // 0..63
  int achunk = (tid & 3) * 8;     // 0,8,16,24 (bf16 elements)

  // per-thread source row for the A tile (k-invariant)
  int m = m0 + arow;
  bool valid = (m < M);
  long srcm = m;
  if (T > 0 && shift != 0) {
    int t  = m % T;
    int ts = t + shift;
    valid  = valid && (ts >= 0) && (ts < T);
    srcm   = (long)m + shift;
  }
  if (!valid) srcm = 0;   // clamped safe address; zero-fixed below

  const unsigned short* gA  = A + (size_t)srcm * K + achunk;
  const unsigned short* gW0 = W + (size_t)(n0 + arow) * K + achunk;
  const unsigned short* gW1 = W + (size_t)(n0 + arow + 64) * K + achunk;
  unsigned lA[2], lW0[2], lW1[2];
#pragma unroll
  for (int bfi = 0; bfi < 2; bfi++) {
    lA[bfi]  = lds_addr32(&sA[bfi][arow * 32 + achunk]);
    lW0[bfi] = lds_addr32(&sW[bfi][arow * 32 + achunk]);
    lW1[bfi] = lds_addr32(&sW[bfi][(arow + 64) * 32 + achunk]);
  }

  // block-uniform: does this block contain rows needing zero fill?
  bool edge = (m0 + 64 > M) ||
              (T > 0 && shift != 0 && ((m0 % T) == 0 || (m0 % T) + 64 == T));

  int nk = K / 32;
  auto issue = [&](int i) {
    int b = i & 1;
    size_t koff = (size_t)i * 32;   // bf16 elements
    async_ld_b128(lA[b],  (unsigned long long)(uintptr_t)(gA  + koff));
    async_ld_b128(lW0[b], (unsigned long long)(uintptr_t)(gW0 + koff));
    async_ld_b128(lW1[b], (unsigned long long)(uintptr_t)(gW1 + koff));
  };

  v8f acc[2][2] = {};
  issue(0);
  for (int i = 0; i < nk; i++) {
    bool hasnext = (i + 1 < nk);
    if (hasnext) {
      issue(i + 1);
      asm volatile("s_wait_asynccnt 3" ::: "memory");
    } else {
      asm volatile("s_wait_asynccnt 0" ::: "memory");
    }
    __syncthreads();                 // tile i resident in LDS across all waves
    int b = i & 1;
    if (edge) {
      if (!valid) *(uint4*)&sA[b][arow * 32 + achunk] = make_uint4(0u,0u,0u,0u);
      __syncthreads();
    }

    Frag af[2], bf[2];
#pragma unroll
    for (int ii = 0; ii < 2; ii++) {
      int r = wm * 32 + ii * 16 + ln16;
      af[ii].q[0] = *(const uint4*)&sA[b][r * 32 + klo];
      af[ii].q[1] = *(const uint4*)&sA[b][r * 32 + 16 + klo];
    }
#pragma unroll
    for (int jj = 0; jj < 2; jj++) {
      int c = wn * 32 + jj * 16 + ln16;
      bf[jj].q[0] = *(const uint4*)&sW[b][c * 32 + klo];
      bf[jj].q[1] = *(const uint4*)&sW[b][c * 32 + 16 + klo];
    }
#pragma unroll
    for (int ii = 0; ii < 2; ii++)
#pragma unroll
      for (int jj = 0; jj < 2; jj++)
        acc[ii][jj] = wmma_bf16(af[ii], bf[jj], acc[ii][jj]);
    __syncthreads();                 // buf reusable for tile i+2
  }

#pragma unroll
  for (int i = 0; i < 2; i++) {
#pragma unroll
    for (int j = 0; j < 2; j++) {
      int n = n0 + wn * 32 + j * 16 + ln16;
      float bv = bias ? bias[n] : 0.f;
#pragma unroll
      for (int e = 0; e < 8; e++) {
        int mm = m0 + wm * 32 + i * 16 + e + 8 * hf;
        if (mm < M) {
          float val = acc[i][j][e] + bv;
          size_t idx = (size_t)mm * N + n;
          if (accum) C[idx] += val; else C[idx] = val;
        }
      }
    }
  }
}

// ---------------------------------------------------------------------------
// Fused rel-pos flash attention.
// rel_shift folded: scores[t,s] = qu[t].k[s] + qv[t].p[s - t + (T-1)]
// Each wave: 16 t-rows; loop over s in steps of 32; per step:
//   AC 6 wmma, BD (Lo/Hi) 12 wmma, LDS diagonal combine, online softmax,
//   P transposed through LDS into A-fragments, P@V 6 wmma.
// qu/qv are pre-scaled by 1/sqrt(DH) at pack time.
// ---------------------------------------------------------------------------
__global__ __launch_bounds__(256) void flash_relpos_kernel(
    const unsigned short* __restrict__ QU, const unsigned short* __restrict__ QV,
    const unsigned short* __restrict__ Kp, const unsigned short* __restrict__ VT,
    const unsigned short* __restrict__ Pp, float* __restrict__ O)
{
  __shared__ float sLo[8][16][16];
  __shared__ float sHi[8][16][16];
  __shared__ __align__(16) unsigned short sP[8][16][32];

  int tid  = threadIdx.x;
  int lane = tid & 31;
  int wave = tid >> 5;
  int ln16 = lane & 15;
  int hf   = (lane >> 4) & 1;
  int klo  = hf * 8;

  int nTB = Tc / 128;
  int bh  = blockIdx.x / nTB;
  int tb  = blockIdx.x % nTB;
  int h   = bh % Hn;
  int bb  = bh / Hn;
  int t0  = tb * 128 + wave * 16;

  // preload q+u / q+v A-fragments (K = 96 -> 3 frags each)
  Frag qu[3], qv[3];
#pragma unroll
  for (int kk = 0; kk < 3; kk++) {
    size_t qidx = (((size_t)bh * Tc) + t0 + ln16) * DHc + kk * 32 + klo;
    qu[kk].q[0] = ldg16(QU + qidx);  qu[kk].q[1] = ldg16(QU + qidx + 16);
    qv[kk].q[0] = ldg16(QV + qidx);  qv[kk].q[1] = ldg16(QV + qidx + 16);
  }

  v8f   oacc[6] = {};
  float mrow[8], lrow[8];
#pragma unroll
  for (int e = 0; e < 8; e++) { mrow[e] = -3.0e38f; lrow[e] = 0.f; }

  for (int s0 = 0; s0 < Tc; s0 += 32) {
    v8f accS[2];
#pragma unroll
    for (int sci = 0; sci < 2; sci++) {
      int sc = s0 + sci * 16;
      int L0 = sc - t0 + (Tc - 1);
      v8f c = {}, cLo = {}, cHi = {};
#pragma unroll
      for (int kk = 0; kk < 3; kk++) {
        // K b-fragment (col = s index)
        Frag kf;
        size_t kidx = (((size_t)bh * Tc) + sc + ln16) * DHc + kk * 32 + klo;
        kf.q[0] = ldg16(Kp + kidx);  kf.q[1] = ldg16(Kp + kidx + 16);
        c = wmma_bf16(qu[kk], kf, c);
        // P b-fragments: Hi covers l = L0..L0+15, Lo covers l = L0-16..L0-1
        int lhi = L0 + ln16, llo = lhi - 16;
        Frag ph, pl;
        if (lhi >= 0 && lhi < Lc) {
          size_t pidx = ((size_t)h * Lc + lhi) * DHc + kk * 32 + klo;
          ph.q[0] = ldg16(Pp + pidx);  ph.q[1] = ldg16(Pp + pidx + 16);
        } else { ph.q[0] = ph.q[1] = make_uint4(0u,0u,0u,0u); }
        if (llo >= 0 && llo < Lc) {
          size_t pidx = ((size_t)h * Lc + llo) * DHc + kk * 32 + klo;
          pl.q[0] = ldg16(Pp + pidx);  pl.q[1] = ldg16(Pp + pidx + 16);
        } else { pl.q[0] = pl.q[1] = make_uint4(0u,0u,0u,0u); }
        cHi = wmma_bf16(qv[kk], ph, cHi);
        cLo = wmma_bf16(qv[kk], pl, cLo);
      }
      // rel-shift diagonal combine via per-wave LDS (DS ops in-order per wave)
#pragma unroll
      for (int e = 0; e < 8; e++) {
        int tm = e + 8 * hf;
        sHi[wave][tm][ln16] = cHi[e];
        sLo[wave][tm][ln16] = cLo[e];
      }
#pragma unroll
      for (int e = 0; e < 8; e++) {
        int tm  = e + 8 * hf;
        int dlt = ln16 - tm;
        float addv = (dlt >= 0) ? sHi[wave][tm][dlt] : sLo[wave][tm][dlt + 16];
        c[e] += addv;
      }
      accS[sci] = c;
    }

    // online softmax (row reductions across 16-lane halves via shfl_xor)
    float scl[8];
#pragma unroll
    for (int e = 0; e < 8; e++) {
      float v = fmaxf(accS[0][e], accS[1][e]);
      v = fmaxf(v, __shfl_xor(v, 1, 32));
      v = fmaxf(v, __shfl_xor(v, 2, 32));
      v = fmaxf(v, __shfl_xor(v, 4, 32));
      v = fmaxf(v, __shfl_xor(v, 8, 32));
      float nm = fmaxf(mrow[e], v);
      scl[e] = __expf(mrow[e] - nm);
      mrow[e] = nm;
    }
#pragma unroll
    for (int e = 0; e < 8; e++) {
      float p0 = __expf(accS[0][e] - mrow[e]);
      float p1 = __expf(accS[1][e] - mrow[e]);
      int tm = e + 8 * hf;
      sP[wave][tm][ln16]      = f2bf(p0);
      sP[wave][tm][16 + ln16] = f2bf(p1);
      float rs = p0 + p1;
      rs += __shfl_xor(rs, 1, 32);
      rs += __shfl_xor(rs, 2, 32);
      rs += __shfl_xor(rs, 4, 32);
      rs += __shfl_xor(rs, 8, 32);
      lrow[e] = lrow[e] * scl[e] + rs;
    }
#pragma unroll
    for (int n = 0; n < 6; n++)
#pragma unroll
      for (int e = 0; e < 8; e++) oacc[n][e] *= scl[e];

    // P (C-layout) -> A-fragment via LDS, then P @ V
    Frag pf;
    pf.q[0] = *(const uint4*)&sP[wave][ln16][klo];
    pf.q[1] = *(const uint4*)&sP[wave][ln16][16 + klo];
#pragma unroll
    for (int n = 0; n < 6; n++) {
      Frag vf;
      int d = n * 16 + ln16;
      size_t vidx = ((size_t)bh * DHc + d) * Tc + s0 + klo;
      vf.q[0] = ldg16(VT + vidx);  vf.q[1] = ldg16(VT + vidx + 16);
      oacc[n] = wmma_bf16(pf, vf, oacc[n]);
    }
  }

  // normalize and write O (M, D) f32 at head offset
#pragma unroll
  for (int e = 0; e < 8; e++) lrow[e] = 1.f / lrow[e];
#pragma unroll
  for (int n = 0; n < 6; n++)
#pragma unroll
    for (int e = 0; e < 8; e++) {
      int tm = e + 8 * hf;
      O[((size_t)bb * Tc + t0 + tm) * Dc + h * DHc + n * 16 + ln16] =
          oacc[n][e] * lrow[e];
    }
}

// ---------------------------------------------------------------------------
// LayerNorm over D=384, writes bf16 and/or f32
// ---------------------------------------------------------------------------
__global__ __launch_bounds__(128) void layernorm_kernel(
    const float* __restrict__ x, const float* __restrict__ g,
    const float* __restrict__ bb, unsigned short* __restrict__ obf,
    float* __restrict__ of32)
{
  int row = blockIdx.x, tid = threadIdx.x;
  const float* xr = x + (size_t)row * Dc;
  float v0 = xr[tid], v1 = xr[tid + 128], v2 = xr[tid + 256];
  float s = v0 + v1 + v2;
#pragma unroll
  for (int m = 16; m >= 1; m >>= 1) s += __shfl_xor(s, m, 32);
  __shared__ float r1[4], r2[4];
  int wv = tid >> 5;
  if ((tid & 31) == 0) r1[wv] = s;
  __syncthreads();
  float mean = (r1[0] + r1[1] + r1[2] + r1[3]) * (1.f / Dc);
  float d0 = v0 - mean, d1 = v1 - mean, d2 = v2 - mean;
  float q = d0 * d0 + d1 * d1 + d2 * d2;
#pragma unroll
  for (int m = 16; m >= 1; m >>= 1) q += __shfl_xor(q, m, 32);
  if ((tid & 31) == 0) r2[wv] = q;
  __syncthreads();
  float var = (r2[0] + r2[1] + r2[2] + r2[3]) * (1.f / Dc);
  float rstd = rsqrtf(var + 1e-5f);
#pragma unroll
  for (int i = 0; i < 3; i++) {
    int c = tid + i * 128;
    float dv = (i == 0) ? d0 : ((i == 1) ? d1 : d2);
    float y = dv * rstd * g[c] + bb[c];
    if (obf)  obf[(size_t)row * Dc + c] = f2bf(y);
    if (of32) of32[(size_t)row * Dc + c] = y;
  }
}

// ---------------------------------------------------------------------------
// Elementwise / packing kernels
// ---------------------------------------------------------------------------
__global__ void pack_w_kernel(const float* __restrict__ w,
                              unsigned short* __restrict__ out,
                              int OI, int Kc, int j) {
  int i = blockIdx.x * 256 + threadIdx.x;
  if (i < OI) out[i] = f2bf(w[(size_t)i * Kc + j]);
}

__global__ void copy_kernel(const float* __restrict__ in, float* __restrict__ out, int n) {
  int i = blockIdx.x * 256 + threadIdx.x;
  if (i < n) out[i] = in[i];
}

__global__ void f32_to_bf16_kernel(const float* __restrict__ in,
                                   unsigned short* __restrict__ out, int n) {
  int i = blockIdx.x * 256 + threadIdx.x;
  if (i < n) out[i] = f2bf(in[i]);
}

__global__ void relu_pack_kernel(const float* __restrict__ in,
                                 unsigned short* __restrict__ out, int n) {
  int i = blockIdx.x * 256 + threadIdx.x;
  if (i < n) out[i] = f2bf(fmaxf(in[i], 0.f));
}

__global__ void add_scaled_kernel(float* __restrict__ X, const float* __restrict__ Y,
                                  float alpha, int n) {
  int i = blockIdx.x * 256 + threadIdx.x;
  if (i < n) X[i] += alpha * Y[i];
}

__global__ void glu_kernel(const float* __restrict__ C1, float* __restrict__ U, int n) {
  int i = blockIdx.x * 256 + threadIdx.x;
  if (i >= n) return;
  int m = i / Dc, c = i % Dc;
  float a = C1[(size_t)m * (2 * Dc) + c];
  float gg = C1[(size_t)m * (2 * Dc) + Dc + c];
  U[i] = a * (1.f / (1.f + __expf(-gg)));
}

__global__ void dw_bn_silu_kernel(const float* __restrict__ U,
                                  const float* __restrict__ dww,
                                  const float* __restrict__ dwb,
                                  const float* __restrict__ bng,
                                  const float* __restrict__ bnb,
                                  unsigned short* __restrict__ out, int n) {
  int i = blockIdx.x * 256 + threadIdx.x;
  if (i >= n) return;
  int m = i / Dc, c = i % Dc;
  int t = m % Tc;
  float acc = dwb[c];
#pragma unroll
  for (int j = 0; j < 7; j++) {
    int tt = t + j - 3;
    if (tt >= 0 && tt < Tc)
      acc += U[(size_t)(m + j - 3) * Dc + c] * dww[c * 7 + j];
  }
  float y = acc * (bng[c] * rsqrtf(1.f + 1e-5f)) + bnb[c];
  y = y * (1.f / (1.f + __expf(-y)));
  out[i] = f2bf(y);
}

__global__ void pack_qkv_kernel(const float* __restrict__ Q,
                                const float* __restrict__ Kf,
                                const float* __restrict__ Vf,
                                const float* __restrict__ pbu,
                                const float* __restrict__ pbv,
                                unsigned short* __restrict__ QU,
                                unsigned short* __restrict__ QV,
                                unsigned short* __restrict__ Kp,
                                unsigned short* __restrict__ VT) {
  int i = blockIdx.x * 256 + threadIdx.x;
  if (i >= Mc * Dc) return;
  int m = i / Dc, c = i % Dc;
  int b = m / Tc, t = m % Tc;
  int h = c / DHc, d = c % DHc;
  const float rs = 0.10206207261596577f;  // 1/sqrt(96)
  size_t dst = (((size_t)b * Hn + h) * Tc + t) * DHc + d;
  float qv0 = Q[i];
  QU[dst] = f2bf((qv0 + pbu[h * DHc + d]) * rs);
  QV[dst] = f2bf((qv0 + pbv[h * DHc + d]) * rs);
  Kp[dst] = f2bf(Kf[i]);
  VT[(((size_t)b * Hn + h) * DHc + d) * Tc + t] = f2bf(Vf[i]);
}

__global__ void pack_p_kernel(const float* __restrict__ Pf,
                              unsigned short* __restrict__ Pp) {
  int i = blockIdx.x * 256 + threadIdx.x;
  if (i >= Lc * Dc) return;
  int l = i / Dc, c = i % Dc;
  int h = c / DHc, d = c % DHc;
  Pp[((size_t)h * Lc + l) * DHc + d] = f2bf(Pf[i]);
}

// ---------------------------------------------------------------------------
// Host orchestration
// ---------------------------------------------------------------------------
extern "C" void kernel_launch(void* const* d_in, const int* in_sizes, int n_in,
                              void* d_out, int out_size, void* d_ws, size_t ws_size,
                              hipStream_t stream) {
  (void)in_sizes; (void)n_in; (void)out_size; (void)ws_size;
  const float* hs   = (const float*)d_in[0];
  const float* pemb = (const float*)d_in[1];
  const float* wq = (const float*)d_in[2];  const float* bq = (const float*)d_in[3];
  const float* wk = (const float*)d_in[4];  const float* bk = (const float*)d_in[5];
  const float* wv = (const float*)d_in[6];  const float* bvb = (const float*)d_in[7];
  const float* wo = (const float*)d_in[8];  const float* bo = (const float*)d_in[9];
  const float* wpos = (const float*)d_in[10];
  const float* pbu = (const float*)d_in[11];
  const float* pbv = (const float*)d_in[12];
  const float* ffm_w1 = (const float*)d_in[13]; const float* ffm_b1 = (const float*)d_in[14];
  const float* ffm_w2 = (const float*)d_in[15]; const float* ffm_b2 = (const float*)d_in[16];
  const float* ff_w1  = (const float*)d_in[17]; const float* ff_b1  = (const float*)d_in[18];
  const float* ff_w2  = (const float*)d_in[19]; const float* ff_b2  = (const float*)d_in[20];
  const float* pw1_w = (const float*)d_in[21]; const float* pw1_b = (const float*)d_in[22];
  const float* dww   = (const float*)d_in[23]; const float* dwb   = (const float*)d_in[24];
  const float* bng   = (const float*)d_in[25]; const float* bnb   = (const float*)d_in[26];
  const float* pw2_w = (const float*)d_in[27]; const float* pw2_b = (const float*)d_in[28];
  const float* ln_ffm_g  = (const float*)d_in[29]; const float* ln_ffm_b  = (const float*)d_in[30];
  const float* ln_attn_g = (const float*)d_in[31]; const float* ln_attn_b = (const float*)d_in[32];
  const float* ln_conv_g = (const float*)d_in[33]; const float* ln_conv_b = (const float*)d_in[34];
  const float* ln_ff_g   = (const float*)d_in[35]; const float* ln_ff_b   = (const float*)d_in[36];
  const float* ln_fin_g  = (const float*)d_in[37]; const float* ln_fin_b  = (const float*)d_in[38];

  char* cur = (char*)d_ws;
  auto alloc = [&](size_t bytes) -> void* {
    void* r = (void*)cur; cur += (bytes + 255) & ~(size_t)255; return r;
  };
  float* X            = (float*)alloc((size_t)Mc * Dc * 4);
  unsigned short* Hbf = (unsigned short*)alloc((size_t)Mc * Dc * 2);
  float* Yb           = (float*)alloc((size_t)Mc * Dc * 4);
  float* RA           = (float*)alloc((size_t)Mc * Fc * 4);           // G1 | Q/K/V | C1+U
  unsigned short* RB  = (unsigned short*)alloc((size_t)Mc * Fc * 2);  // G1bf | QU/QV/K/VT | Hc
  float* Of           = (float*)alloc((size_t)Mc * Dc * 4);
  float* Pf           = (float*)alloc((size_t)Lc * Dc * 4);
  unsigned short* Ppb = (unsigned short*)alloc((size_t)Hn * Lc * DHc * 2);
  auto wal = [&](size_t elems) { return (unsigned short*)alloc(elems * 2); };
  unsigned short* wqb = wal((size_t)Dc * Dc);
  unsigned short* wkb = wal((size_t)Dc * Dc);
  unsigned short* wvb = wal((size_t)Dc * Dc);
  unsigned short* wob = wal((size_t)Dc * Dc);
  unsigned short* wposb = wal((size_t)Dc * Dc);
  unsigned short* ffm1b[3]; unsigned short* ffm2b[3];
  unsigned short* ff1b[3];  unsigned short* ff2b[3];
  for (int j = 0; j < 3; j++) {
    ffm1b[j] = wal((size_t)Fc * Dc); ffm2b[j] = wal((size_t)Dc * Fc);
    ff1b[j]  = wal((size_t)Fc * Dc); ff2b[j]  = wal((size_t)Dc * Fc);
  }
  unsigned short* pw1b = wal((size_t)2 * Dc * Dc);
  unsigned short* pw2b = wal((size_t)Dc * Dc);

  float* G1  = RA;
  float* Qf  = RA;
  float* Kf2 = RA + (size_t)Mc * Dc;
  float* Vf2 = RA + (size_t)2 * Mc * Dc;
  float* C1  = RA;
  float* Ug  = RA + (size_t)Mc * 2 * Dc;
  unsigned short* G1bf = RB;
  unsigned short* QUb = RB;
  unsigned short* QVb = RB + (size_t)Mc * Dc;
  unsigned short* Kpb = RB + (size_t)2 * Mc * Dc;
  unsigned short* VTb = RB + (size_t)3 * Mc * Dc;
  unsigned short* Hcb = RB;

  auto ew = [](int n) { return dim3((unsigned)((n + 255) / 256)); };

  // ---- pack weights to bf16 ----
  pack_w_kernel<<<ew(Dc*Dc),256,0,stream>>>(wq,   wqb,   Dc*Dc, 1, 0);
  pack_w_kernel<<<ew(Dc*Dc),256,0,stream>>>(wk,   wkb,   Dc*Dc, 1, 0);
  pack_w_kernel<<<ew(Dc*Dc),256,0,stream>>>(wv,   wvb,   Dc*Dc, 1, 0);
  pack_w_kernel<<<ew(Dc*Dc),256,0,stream>>>(wo,   wob,   Dc*Dc, 1, 0);
  pack_w_kernel<<<ew(Dc*Dc),256,0,stream>>>(wpos, wposb, Dc*Dc, 1, 0);
  for (int j = 0; j < 3; j++) {
    pack_w_kernel<<<ew(Fc*Dc),256,0,stream>>>(ffm_w1, ffm1b[j], Fc*Dc, 3, j);
    pack_w_kernel<<<ew(Dc*Fc),256,0,stream>>>(ffm_w2, ffm2b[j], Dc*Fc, 3, j);
    pack_w_kernel<<<ew(Fc*Dc),256,0,stream>>>(ff_w1,  ff1b[j],  Fc*Dc, 3, j);
    pack_w_kernel<<<ew(Dc*Fc),256,0,stream>>>(ff_w2,  ff2b[j],  Dc*Fc, 3, j);
  }
  pack_w_kernel<<<ew(2*Dc*Dc),256,0,stream>>>(pw1_w, pw1b, 2*Dc*Dc, 1, 0);
  pack_w_kernel<<<ew(Dc*Dc),256,0,stream>>>(pw2_w, pw2b, Dc*Dc, 1, 0);

  copy_kernel<<<ew(Mc*Dc),256,0,stream>>>(hs, X, Mc*Dc);

  auto gemm = [&](const unsigned short* A, const unsigned short* Wt, const float* bias,
                  float* Cc, int Mm, int Nn, int Kk, int Tt, int shift, int accum) {
    dim3 g((unsigned)((Mm + 63) / 64), (unsigned)(Nn / 128));
    gemm_bf16_kernel<<<g, 256, 0, stream>>>(A, Wt, bias, Cc, Mm, Nn, Kk, Tt, shift, accum);
  };

  auto ffn_block = [&](const float* lng, const float* lnb,
                       unsigned short* const* w1b3, const float* b1,
                       unsigned short* const* w2b3, const float* b2) {
    layernorm_kernel<<<Mc,128,0,stream>>>(X, lng, lnb, Hbf, nullptr);
    gemm(Hbf, w1b3[0], b1,      G1, Mc, Fc, Dc, Tc, -1, 0);
    gemm(Hbf, w1b3[1], nullptr, G1, Mc, Fc, Dc, Tc,  0, 1);
    gemm(Hbf, w1b3[2], nullptr, G1, Mc, Fc, Dc, Tc,  1, 1);
    relu_pack_kernel<<<ew(Mc*Fc),256,0,stream>>>(G1, G1bf, Mc*Fc);
    gemm(G1bf, w2b3[0], b2,      Yb, Mc, Dc, Fc, Tc, -1, 0);
    gemm(G1bf, w2b3[1], nullptr, Yb, Mc, Dc, Fc, Tc,  0, 1);
    gemm(G1bf, w2b3[2], nullptr, Yb, Mc, Dc, Fc, Tc,  1, 1);
    add_scaled_kernel<<<ew(Mc*Dc),256,0,stream>>>(X, Yb, 0.5f, Mc*Dc);
  };

  // ---- macaron FFN ----
  ffn_block(ln_ffm_g, ln_ffm_b, ffm1b, ffm_b1, ffm2b, ffm_b2);

  // ---- rel-pos multi-head self-attention ----
  layernorm_kernel<<<Mc,128,0,stream>>>(X, ln_attn_g, ln_attn_b, Hbf, nullptr);
  gemm(Hbf, wqb, bq,  Qf,  Mc, Dc, Dc, 0, 0, 0);
  gemm(Hbf, wkb, bk,  Kf2, Mc, Dc, Dc, 0, 0, 0);
  gemm(Hbf, wvb, bvb, Vf2, Mc, Dc, Dc, 0, 0, 0);
  f32_to_bf16_kernel<<<ew(Lc*Dc),256,0,stream>>>(pemb, Hbf, Lc*Dc);  // reuse Hbf
  gemm(Hbf, wposb, nullptr, Pf, Lc, Dc, Dc, 0, 0, 0);
  pack_qkv_kernel<<<ew(Mc*Dc),256,0,stream>>>(Qf, Kf2, Vf2, pbu, pbv, QUb, QVb, Kpb, VTb);
  pack_p_kernel<<<ew(Lc*Dc),256,0,stream>>>(Pf, Ppb);
  flash_relpos_kernel<<<Bc*Hn*(Tc/128), 256, 0, stream>>>(QUb, QVb, Kpb, VTb, Ppb, Of);
  f32_to_bf16_kernel<<<ew(Mc*Dc),256,0,stream>>>(Of, Hbf, Mc*Dc);    // reuse Hbf
  gemm(Hbf, wob, bo, Yb, Mc, Dc, Dc, 0, 0, 0);
  add_scaled_kernel<<<ew(Mc*Dc),256,0,stream>>>(X, Yb, 1.0f, Mc*Dc);

  // ---- convolution module ----
  layernorm_kernel<<<Mc,128,0,stream>>>(X, ln_conv_g, ln_conv_b, Hbf, nullptr);
  gemm(Hbf, pw1b, pw1_b, C1, Mc, 2*Dc, Dc, 0, 0, 0);
  glu_kernel<<<ew(Mc*Dc),256,0,stream>>>(C1, Ug, Mc*Dc);
  dw_bn_silu_kernel<<<ew(Mc*Dc),256,0,stream>>>(Ug, dww, dwb, bng, bnb, Hcb, Mc*Dc);
  gemm(Hcb, pw2b, pw2_b, Yb, Mc, Dc, Dc, 0, 0, 0);
  add_scaled_kernel<<<ew(Mc*Dc),256,0,stream>>>(X, Yb, 1.0f, Mc*Dc);

  // ---- final FFN ----
  ffn_block(ln_ff_g, ln_ff_b, ff1b, ff_b1, ff2b, ff_b2);

  // ---- final layer norm -> d_out (f32) ----
  layernorm_kernel<<<Mc,128,0,stream>>>(X, ln_fin_g, ln_fin_b, nullptr, (float*)d_out);
}
